// CrossAttn_25847113188077
// MI455X (gfx1250) — matrix operand order repaired and checked by reading, not compiled, not run on hardware
//
#include <hip/hip_runtime.h>

typedef _Float16 half_t;
typedef __attribute__((ext_vector_type(16))) _Float16 v16h;
typedef __attribute__((ext_vector_type(8)))  _Float16 v8h;
typedef __attribute__((ext_vector_type(4)))  _Float16 v4h;
typedef __attribute__((ext_vector_type(8)))  float    v8f;

#define DIMC  768
#define NQKV  2304
#define HEADS 12
#define HD    64
#define SEQ   1024
#define BATCH 4
#define TOK   (BATCH * SEQ)     // 4096 tokens per stream
#define SEQ2  2048              // concatenated sequence
#define BH    (BATCH * HEADS)   // 48

#define WMMA(a, b, c) \
  __builtin_amdgcn_wmma_f32_16x16x32_f16(false, (a), false, (b), (short)0, (c), false, false)

__device__ __forceinline__ v8f zero8() {
  v8f z;
#pragma unroll
  for (int i = 0; i < 8; ++i) z[i] = 0.0f;
  return z;
}

__device__ __forceinline__ v16h frag_cat(const half_t* p0, const half_t* p1) {
  v8h a = *(const v8h*)p0;
  v8h b = *(const v8h*)p1;
  v16h r;
#pragma unroll
  for (int i = 0; i < 8; ++i) { r[i] = a[i]; r[i + 8] = b[i]; }
  return r;
}

// ---------------------------------------------------------------------------
// C[M,N] = A[M,K](f32) * B[N,K]^T(f32) + bias[N], stored f16 (QKV path).
// 128x128 tile, 8 waves, wave tile 32x64. N,K compile-time.
// ---------------------------------------------------------------------------
template <int N, int K>
__global__ void __launch_bounds__(256)
gemm_f32_nt(const float* __restrict__ A, const float* __restrict__ B,
            const float* __restrict__ bias, half_t* __restrict__ C) {
  __shared__ __align__(16) half_t As[128 * 40];
  __shared__ __align__(16) half_t Bs[128 * 40];
  const int tid = threadIdx.x, lane = tid & 31, wave = tid >> 5;
  const int tn = blockIdx.x * 128, tm = blockIdx.y * 128;
  const int wm = (wave >> 1) * 32, wn = (wave & 1) * 64;

  // loop-invariant staging pointers
  const float *pa[4], *pb[4];
  half_t *sa[4], *sb[4];
#pragma unroll
  for (int it = 0; it < 4; ++it) {
    int id = tid + it * 256;             // 0..1023 float4 chunks
    int r = id >> 3, c = (id & 7) << 2;  // row 0..127, k 0..28
    pa[it] = A + (tm + r) * K + c;
    pb[it] = B + (tn + r) * K + c;
    sa[it] = &As[r * 40 + c];
    sb[it] = &Bs[r * 40 + c];
  }
  // loop-invariant fragment pointers
  const half_t *afp[2], *bfp[4];
#pragma unroll
  for (int i = 0; i < 2; ++i)
    afp[i] = &As[(wm + 16 * i + (lane & 15)) * 40 + ((lane >> 4) << 3)];
#pragma unroll
  for (int j = 0; j < 4; ++j)
    bfp[j] = &Bs[(wn + 16 * j + (lane & 15)) * 40 + ((lane >> 4) << 3)];

  v8f acc[2][4];
#pragma unroll
  for (int i = 0; i < 2; ++i)
#pragma unroll
    for (int j = 0; j < 4; ++j) acc[i][j] = zero8();

#pragma unroll 2
  for (int k0 = 0; k0 < K; k0 += 32) {
    __syncthreads();
#pragma unroll
    for (int it = 0; it < 4; ++it) {
      float4 da = *(const float4*)(pa[it] + k0);
      float4 db = *(const float4*)(pb[it] + k0);
      v4h ha, hb;
      ha[0] = (half_t)da.x; ha[1] = (half_t)da.y; ha[2] = (half_t)da.z; ha[3] = (half_t)da.w;
      hb[0] = (half_t)db.x; hb[1] = (half_t)db.y; hb[2] = (half_t)db.z; hb[3] = (half_t)db.w;
      *(v4h*)sa[it] = ha;
      *(v4h*)sb[it] = hb;
    }
    __syncthreads();
    v16h af[2], bf[4];
#pragma unroll
    for (int i = 0; i < 2; ++i) af[i] = frag_cat(afp[i], afp[i] + 16);
#pragma unroll
    for (int j = 0; j < 4; ++j) bf[j] = frag_cat(bfp[j], bfp[j] + 16);
#pragma unroll
    for (int i = 0; i < 2; ++i)
#pragma unroll
      for (int j = 0; j < 4; ++j) acc[i][j] = WMMA(af[i], bf[j], acc[i][j]);
  }

  const int rsel = (lane >> 4) << 3, csel = lane & 15;
  half_t* cp = C + (tm + wm + rsel) * N + tn + wn + csel;
  float bv[4];
#pragma unroll
  for (int j = 0; j < 4; ++j) bv[j] = bias[tn + wn + j * 16 + csel];
#pragma unroll
  for (int i = 0; i < 2; ++i)
#pragma unroll
    for (int j = 0; j < 4; ++j)
#pragma unroll
      for (int v = 0; v < 8; ++v)
        cp[(i * 16 + v) * N + j * 16] = (half_t)(acc[i][j][v] + bv[j]);
}

// ---------------------------------------------------------------------------
// C[M,N] = A[M,K](f16) * B[N,K]^T(f32) + bias[N], stored f32 (proj path).
// ---------------------------------------------------------------------------
template <int N, int K>
__global__ void __launch_bounds__(256)
gemm_f16_nt(const half_t* __restrict__ A, const float* __restrict__ B,
            const float* __restrict__ bias, float* __restrict__ C) {
  __shared__ __align__(16) half_t As[128 * 40];
  __shared__ __align__(16) half_t Bs[128 * 40];
  const int tid = threadIdx.x, lane = tid & 31, wave = tid >> 5;
  const int tn = blockIdx.x * 128, tm = blockIdx.y * 128;
  const int wm = (wave >> 1) * 32, wn = (wave & 1) * 64;

  const half_t* pa[2];
  half_t* sa[2];
#pragma unroll
  for (int it = 0; it < 2; ++it) {
    int id = tid + it * 256;             // 0..511 v8h chunks
    int r = id >> 2, c = (id & 3) << 3;
    pa[it] = A + (tm + r) * K + c;
    sa[it] = &As[r * 40 + c];
  }
  const float* pb[4];
  half_t* sb[4];
#pragma unroll
  for (int it = 0; it < 4; ++it) {
    int id = tid + it * 256;
    int r = id >> 3, c = (id & 7) << 2;
    pb[it] = B + (tn + r) * K + c;
    sb[it] = &Bs[r * 40 + c];
  }
  const half_t *afp[2], *bfp[4];
#pragma unroll
  for (int i = 0; i < 2; ++i)
    afp[i] = &As[(wm + 16 * i + (lane & 15)) * 40 + ((lane >> 4) << 3)];
#pragma unroll
  for (int j = 0; j < 4; ++j)
    bfp[j] = &Bs[(wn + 16 * j + (lane & 15)) * 40 + ((lane >> 4) << 3)];

  v8f acc[2][4];
#pragma unroll
  for (int i = 0; i < 2; ++i)
#pragma unroll
    for (int j = 0; j < 4; ++j) acc[i][j] = zero8();

#pragma unroll 2
  for (int k0 = 0; k0 < K; k0 += 32) {
    __syncthreads();
#pragma unroll
    for (int it = 0; it < 2; ++it)
      *(v8h*)sa[it] = *(const v8h*)(pa[it] + k0);
#pragma unroll
    for (int it = 0; it < 4; ++it) {
      float4 db = *(const float4*)(pb[it] + k0);
      v4h hb;
      hb[0] = (half_t)db.x; hb[1] = (half_t)db.y; hb[2] = (half_t)db.z; hb[3] = (half_t)db.w;
      *(v4h*)sb[it] = hb;
    }
    __syncthreads();
    v16h af[2], bf[4];
#pragma unroll
    for (int i = 0; i < 2; ++i) af[i] = frag_cat(afp[i], afp[i] + 16);
#pragma unroll
    for (int j = 0; j < 4; ++j) bf[j] = frag_cat(bfp[j], bfp[j] + 16);
#pragma unroll
    for (int i = 0; i < 2; ++i)
#pragma unroll
      for (int j = 0; j < 4; ++j) acc[i][j] = WMMA(af[i], bf[j], acc[i][j]);
  }

  const int rsel = (lane >> 4) << 3, csel = lane & 15;
  float* cp = C + (tm + wm + rsel) * N + tn + wn + csel;
  float bv[4];
#pragma unroll
  for (int j = 0; j < 4; ++j) bv[j] = bias[tn + wn + j * 16 + csel];
#pragma unroll
  for (int i = 0; i < 2; ++i)
#pragma unroll
    for (int j = 0; j < 4; ++j)
#pragma unroll
      for (int v = 0; v < 8; ++v)
        cp[(i * 16 + v) * N + j * 16] = acc[i][j][v] + bv[j];
}

// ---------------------------------------------------------------------------
// Per-head LayerNorm on q,k + pack into attention layouts:
//   qU/kU: [bh][seq2][64] f16 (q pre-scaled by 1/sqrt(64)*log2(e))
//   vT:    [bh][64][seq2] f16 (transposed -> NT second GEMM in attention)
// One wave per (stream, token, head) row; 2 elements/lane.
// ---------------------------------------------------------------------------
__global__ void __launch_bounds__(256)
ln_pack(const half_t* __restrict__ qkv_x, const half_t* __restrict__ qkv_y,
        const float* __restrict__ gq_x, const float* __restrict__ bq_x,
        const float* __restrict__ gk_x, const float* __restrict__ bk_x,
        const float* __restrict__ gq_y, const float* __restrict__ bq_y,
        const float* __restrict__ gk_y, const float* __restrict__ bk_y,
        half_t* __restrict__ qU, half_t* __restrict__ kU, half_t* __restrict__ vT) {
  const int lane = threadIdx.x & 31;
  int gw = (blockIdx.x * 256 + threadIdx.x) >> 5;  // 0 .. 2*4096*12-1
  int h = gw % HEADS;
  int t = gw / HEADS;        // 0..8191
  int str = t >> 12;         // stream: 0=x, 1=y
  int tok = t & 4095;
  int b = tok >> 10, s = tok & 1023;
  const half_t* qkv = str ? qkv_y : qkv_x;
  const float* gq = str ? gq_y : gq_x;
  const float* bq = str ? bq_y : bq_x;
  const float* gk = str ? gk_y : gk_x;
  const float* bk = str ? bk_y : bk_x;
  const half_t* row = qkv + tok * NQKV + h * HD;
  const int d0 = lane, d1 = lane + 32;
  float q0 = (float)row[d0],            q1 = (float)row[d1];
  float k0 = (float)row[DIMC + d0],     k1 = (float)row[DIMC + d1];
  float v0 = (float)row[2 * DIMC + d0], v1 = (float)row[2 * DIMC + d1];
  float qs = q0 + q1, qq = q0 * q0 + q1 * q1;
  float ks = k0 + k1, kk = k0 * k0 + k1 * k1;
#pragma unroll
  for (int msk = 1; msk < 32; msk <<= 1) {
    qs += __shfl_xor(qs, msk, 32);  qq += __shfl_xor(qq, msk, 32);
    ks += __shfl_xor(ks, msk, 32);  kk += __shfl_xor(kk, msk, 32);
  }
  float qmu = qs * (1.f / 64.f), kmu = ks * (1.f / 64.f);
  float qrs = rsqrtf(fmaxf(qq * (1.f / 64.f) - qmu * qmu, 0.f) + 1e-5f);
  float krs = rsqrtf(fmaxf(kk * (1.f / 64.f) - kmu * kmu, 0.f) + 1e-5f);
  const float qscale = 0.125f * 1.4426950408889634f;  // 1/sqrt(HD) * log2(e)
  float qn0 = ((q0 - qmu) * qrs * gq[d0] + bq[d0]) * qscale;
  float qn1 = ((q1 - qmu) * qrs * gq[d1] + bq[d1]) * qscale;
  float kn0 = (k0 - kmu) * krs * gk[d0] + bk[d0];
  float kn1 = (k1 - kmu) * krs * gk[d1] + bk[d1];
  int bh = b * HEADS + h;
  int su = s + (str << 10);
  half_t* qo = qU + (bh * SEQ2 + su) * HD;
  half_t* ko = kU + (bh * SEQ2 + su) * HD;
  qo[d0] = (half_t)qn0;  qo[d1] = (half_t)qn1;
  ko[d0] = (half_t)kn0;  ko[d1] = (half_t)kn1;
  half_t* vo = vT + bh * HD * SEQ2 + su;
  vo[d0 * SEQ2] = (half_t)v0;
  vo[d1 * SEQ2] = (half_t)v1;
}

// ---------------------------------------------------------------------------
// Flash attention: 8 waves x 16 query rows = 128-row blocks; key blocks of 64.
// S = Q*K^T (exp2 domain), online softmax, O += P*V via vT. Output -> f16
// token-major buffers for the projection GEMMs.
// ---------------------------------------------------------------------------
__global__ void __launch_bounds__(256)
flash_attn(const half_t* __restrict__ qU, const half_t* __restrict__ kU,
           const half_t* __restrict__ vT,
           half_t* __restrict__ ax, half_t* __restrict__ ay) {
  __shared__ __align__(16) half_t Kt[64 * 72];
  __shared__ __align__(16) half_t Vt[64 * 72];
  __shared__ __align__(16) half_t Pl[8][16 * 72];
  const int tid = threadIdx.x, lane = tid & 31, wave = tid >> 5;
  const int bh = blockIdx.y;
  const int qstart = blockIdx.x * 128;
  const half_t* qb = qU + bh * SEQ2 * HD;
  const half_t* kb = kU + bh * SEQ2 * HD;
  const half_t* vb = vT + bh * HD * SEQ2;

  // Q fragments live in registers for the whole kernel (A layout, d = 0..63)
  const int qrow = qstart + wave * 16 + (lane & 15);
  const int koff = (lane >> 4) << 3;
  const half_t* qp = qb + qrow * HD + koff;
  v16h qf0 = frag_cat(qp, qp + 16);
  v16h qf1 = frag_cat(qp + 32, qp + 48);

  // loop-invariant staging + fragment pointers
  const half_t *pk[2], *pv[2];
  half_t *sk[2], *sv[2];
#pragma unroll
  for (int it = 0; it < 2; ++it) {
    int id = tid + it * 256;             // 0..511 v8h chunks
    int r = id >> 3, c = (id & 7) << 3;  // r 0..63, c 0..56
    pk[it] = kb + r * HD + c;
    pv[it] = vb + r * SEQ2 + c;
    sk[it] = &Kt[r * 72 + c];
    sv[it] = &Vt[r * 72 + c];
  }
  const half_t *kfp[4], *vfp[4], *pfp;
#pragma unroll
  for (int n = 0; n < 4; ++n)
    kfp[n] = &Kt[(n * 16 + (lane & 15)) * 72 + ((lane >> 4) << 3)];
#pragma unroll
  for (int d = 0; d < 4; ++d)
    vfp[d] = &Vt[(d * 16 + (lane & 15)) * 72 + ((lane >> 4) << 3)];
  pfp = &Pl[wave][(lane & 15) * 72 + ((lane >> 4) << 3)];
  half_t* psp = &Pl[wave][((lane >> 4) << 3) * 72 + (lane & 15)];

  v8f o[4];
  float m[8], l[8];
#pragma unroll
  for (int v = 0; v < 8; ++v) { m[v] = -1e30f; l[v] = 0.f; }
#pragma unroll
  for (int d = 0; d < 4; ++d) o[d] = zero8();

  for (int kblk = 0; kblk < SEQ2; kblk += 64) {
    __syncthreads();
#pragma unroll
    for (int it = 0; it < 2; ++it) {
      *(v8h*)sk[it] = *(const v8h*)(pk[it] + kblk * HD);
      *(v8h*)sv[it] = *(const v8h*)(pv[it] + kblk);
    }
    __syncthreads();

    // S = Q*K^T for 16 q-rows x 64 keys (4 fragments)
    v8f s[4];
#pragma unroll
    for (int n = 0; n < 4; ++n) {
      v16h b0 = frag_cat(kfp[n], kfp[n] + 16);           // d 0..31
      v16h b1 = frag_cat(kfp[n] + 32, kfp[n] + 48);      // d 32..63
      v8f z = zero8();
      z = WMMA(qf0, b0, z);
      s[n] = WMMA(qf1, b1, z);
    }

    // Online softmax (rows distributed: vgpr v -> row v + 8*(lane>=16))
    float mn[8];
#pragma unroll
    for (int v = 0; v < 8; ++v) {
      float t = fmaxf(fmaxf(s[0][v], s[1][v]), fmaxf(s[2][v], s[3][v]));
#pragma unroll
      for (int msk = 1; msk < 16; msk <<= 1) t = fmaxf(t, __shfl_xor(t, msk, 32));
      mn[v] = fmaxf(t, m[v]);
    }
    float rs[8];
#pragma unroll
    for (int v = 0; v < 8; ++v) {
      float sc = exp2f(m[v] - mn[v]);
      m[v] = mn[v];
      l[v] *= sc;
      rs[v] = 0.f;
#pragma unroll
      for (int d = 0; d < 4; ++d) o[d][v] = o[d][v] * sc;
    }
#pragma unroll
    for (int n = 0; n < 4; ++n)
#pragma unroll
      for (int v = 0; v < 8; ++v) {
        float p = exp2f(s[n][v] - m[v]);
        rs[v] += p;
        psp[v * 72 + n * 16] = (half_t)p;
      }
#pragma unroll
    for (int v = 0; v < 8; ++v) {
#pragma unroll
      for (int msk = 1; msk < 16; msk <<= 1) rs[v] += __shfl_xor(rs[v], msk, 32);
      l[v] += rs[v];
    }

    // Same-wave cross-lane LDS RAW: wait for DS stores before A-frag reads.
    asm volatile("s_wait_dscnt 0x0" ::: "memory");
    v16h pf0 = frag_cat(pfp, pfp + 16);        // keys 0..31
    v16h pf1 = frag_cat(pfp + 32, pfp + 48);   // keys 32..63
#pragma unroll
    for (int d = 0; d < 4; ++d) {
      o[d] = WMMA(pf0, frag_cat(vfp[d], vfp[d] + 16), o[d]);
      o[d] = WMMA(pf1, frag_cat(vfp[d] + 32, vfp[d] + 48), o[d]);
    }
  }

  // 128-row blocks never straddle the x/y boundary (1024 % 128 == 0)
  const int h = bh % HEADS, b = bh / HEADS;
  const int rsel = (lane >> 4) << 3, csel = lane & 15;
  const bool isx = qstart < SEQ;
  half_t* outb = isx ? (ax + (b * SEQ + qstart) * DIMC)
                     : (ay + (b * SEQ + qstart - SEQ) * DIMC);
  half_t* op = outb + (wave * 16 + rsel) * DIMC + h * HD + csel;
#pragma unroll
  for (int d = 0; d < 4; ++d) {
#pragma unroll
    for (int v = 0; v < 8; ++v)
      op[v * DIMC + d * 16] = (half_t)(o[d][v] / l[v]);
  }
}

// ---------------------------------------------------------------------------
extern "C" void kernel_launch(void* const* d_in, const int* in_sizes, int n_in,
                              void* d_out, int out_size, void* d_ws, size_t ws_size,
                              hipStream_t stream) {
  const float* x       = (const float*)d_in[0];
  const float* y       = (const float*)d_in[1];
  const float* Wqkv_x  = (const float*)d_in[2];
  const float* bqkv_x  = (const float*)d_in[3];
  const float* Wqkv_y  = (const float*)d_in[4];
  const float* bqkv_y  = (const float*)d_in[5];
  const float* Wproj_x = (const float*)d_in[6];
  const float* bproj_x = (const float*)d_in[7];
  const float* Wproj_y = (const float*)d_in[8];
  const float* bproj_y = (const float*)d_in[9];
  const float* gq_x = (const float*)d_in[10];
  const float* bq_x = (const float*)d_in[11];
  const float* gk_x = (const float*)d_in[12];
  const float* bk_x = (const float*)d_in[13];
  const float* gq_y = (const float*)d_in[14];
  const float* bq_y = (const float*)d_in[15];
  const float* gk_y = (const float*)d_in[16];
  const float* bk_y = (const float*)d_in[17];

  half_t* ws = (half_t*)d_ws;
  const size_t QKVSZ = (size_t)TOK * NQKV;        // 9.4M halves each
  const size_t HSZ   = (size_t)BH * SEQ2 * HD;    // 6.3M halves each
  const size_t ASZ   = (size_t)TOK * DIMC;        // 3.1M halves each
  half_t* qkv_x = ws;
  half_t* qkv_y = qkv_x + QKVSZ;
  half_t* qU = qkv_y + QKVSZ;
  half_t* kU = qU + HSZ;
  half_t* vT = kU + HSZ;
  half_t* ax = vT + HSZ;
  half_t* ay = ax + ASZ;
  float* out = (float*)d_out;

  dim3 blk(256);
  gemm_f32_nt<NQKV, DIMC><<<dim3(NQKV / 128, TOK / 128), blk, 0, stream>>>(
      x, Wqkv_x, bqkv_x, qkv_x);
  gemm_f32_nt<NQKV, DIMC><<<dim3(NQKV / 128, TOK / 128), blk, 0, stream>>>(
      y, Wqkv_y, bqkv_y, qkv_y);
  ln_pack<<<dim3((2 * TOK * HEADS) / 8), blk, 0, stream>>>(
      qkv_x, qkv_y, gq_x, bq_x, gk_x, bk_x, gq_y, bq_y, gk_y, bk_y, qU, kU, vT);
  flash_attn<<<dim3(SEQ2 / 128, BH), blk, 0, stream>>>(qU, kU, vT, ax, ay);
  gemm_f16_nt<DIMC, DIMC><<<dim3(DIMC / 128, TOK / 128), blk, 0, stream>>>(
      ax, Wproj_x, bproj_x, out);
  gemm_f16_nt<DIMC, DIMC><<<dim3(DIMC / 128, TOK / 128), blk, 0, stream>>>(
      ay, Wproj_y, bproj_y, out + ASZ);
}
